// AttentionBlock_32349693673845
// MI455X (gfx1250) — matrix operand order repaired and checked by reading, not compiled, not run on hardware
//
#include <hip/hip_runtime.h>
#include <hip/hip_bf16.h>

typedef __attribute__((ext_vector_type(16))) _Float16 v16h;
typedef __attribute__((ext_vector_type(8)))  _Float16 v8h;
typedef __attribute__((ext_vector_type(8)))  float    v8f;
typedef __attribute__((ext_vector_type(4)))  unsigned int u32x4;
typedef __attribute__((ext_vector_type(4)))  int i32x4;
typedef __attribute__((ext_vector_type(8)))  int i32x8;

#define CC 256
#define NN 4096
#define BB 8

#if defined(__has_builtin)
#if __has_builtin(__builtin_amdgcn_tensor_load_to_lds) && \
    __has_builtin(__builtin_amdgcn_s_wait_tensorcnt)
#define HAVE_TDM 1
#endif
#endif
#ifndef HAVE_TDM
#define HAVE_TDM 0
#endif

// ---------------------------------------------------------------- WMMA core
__device__ __forceinline__ v8f wmma16(v16h a, v16h b, v8f c) {
  return __builtin_amdgcn_wmma_f32_16x16x32_f16(false, a, false, b, (short)0, c,
                                                false, false);
}

// A fragment (16x32 f16, ISA 7.12.2): lane<16 holds row r=lane, K {0..7,16..23};
// lane>=16 holds row r=lane-16, K {8..15,24..31}.
__device__ __forceinline__ v16h frag_a(const _Float16* t, int stride, int rb,
                                       int kb, int lane) {
  const int r = rb + (lane & 15);
  const int h = lane >> 4;
  const _Float16* p = t + r * stride + kb;
  v8h lo = *(const v8h*)(p + h * 8);
  v8h hi = *(const v8h*)(p + 16 + h * 8);
  v16h out;
#pragma unroll
  for (int i = 0; i < 8; ++i) { out[i] = lo[i]; out[8 + i] = hi[i]; }
  return out;
}

// B fragment (32x16 f16): lane<16 -> col=lane, K {0..7,16..23}; lane>=16 ->
// col=lane-16, K {8..15,24..31}. Source tile stored [col][k] row-major.
__device__ __forceinline__ v16h frag_b(const _Float16* t, int stride, int cb,
                                       int kb, int lane) {
  const _Float16* p = t + (cb + (lane & 15)) * stride + kb;
  v8h lo = *(const v8h*)(p + (lane >> 4) * 8);
  v8h hi = *(const v8h*)(p + 16 + (lane >> 4) * 8);
  v16h out;
#pragma unroll
  for (int i = 0; i < 8; ++i) { out[i] = lo[i]; out[8 + i] = hi[i]; }
  return out;
}

__device__ __forceinline__ float rmax16(float v) {
#pragma unroll
  for (int m = 1; m < 16; m <<= 1) v = fmaxf(v, __shfl_xor(v, m, 32));
  return v;
}
__device__ __forceinline__ float rsum16(float v) {
#pragma unroll
  for (int m = 1; m < 16; m <<= 1) v += __shfl_xor(v, m, 32);
  return v;
}

#if HAVE_TDM
// 2-D tile DMA: global (row-major, stride_elems f16/row) -> LDS, with the
// descriptor pad fields reproducing the padded LDS row pitch.
// pad_int: interval code (2^(code+1) DWORDs), pad_amt: (amount-1) DWORDs.
__device__ __forceinline__ void tdm_load_2d(unsigned lds_off, const void* gp,
                                            unsigned tile_w, unsigned tile_h,
                                            unsigned stride_elems,
                                            unsigned pad_int,
                                            unsigned pad_amt) {
  const unsigned long long ga = (unsigned long long)(size_t)gp;
  u32x4 g0;
  g0[0] = 1u;                                  // count = 1 valid descriptor
  g0[1] = lds_off;                             // lds_addr (bytes)
  g0[2] = (unsigned)ga;                        // global_addr[31:0]
  g0[3] = (unsigned)((ga >> 32) & 0x01ffffffu) | (2u << 30);  // [56:32]|type=2
  i32x8 g1;
  g1[0] = (int)((1u << 16) |                   // data_size = 2 bytes
                (1u << 20) |                   // pad_enable (LDS row pitch)
                (pad_int << 22) | (pad_amt << 25));
  g1[1] = (int)((stride_elems & 0xffffu) << 16);        // tensor_dim0 lo16
  g1[2] = (int)((stride_elems >> 16) & 0xffffu);        // dim0 hi16, dim1 lo=0
  g1[3] = (int)(16u | ((tile_w & 0xffffu) << 16));      // dim1 hi (=2^20), tile_dim0
  g1[4] = (int)(tile_h & 0xffffu);                      // tile_dim1, tile_dim2=0
  g1[5] = (int)stride_elems;                            // dim0_stride lo32
  g1[6] = 0;                                            // stride hi, dim1_stride
  g1[7] = 0;
  const i32x4 z4 = {0, 0, 0, 0};
#if __clang_major__ >= 23
  const i32x8 z8 = {0, 0, 0, 0, 0, 0, 0, 0};
  __builtin_amdgcn_tensor_load_to_lds(g0, g1, z4, z4, z8, 0);
#else
  __builtin_amdgcn_tensor_load_to_lds(g0, g1, z4, z4, 0);
#endif
}
#endif

// ---------------------------------------------------------------- GroupNorm
__global__ void __launch_bounds__(256) gn_stats_kernel(
    const float* __restrict__ x, float* __restrict__ stats) {
  const int bg = blockIdx.x;  // b*32+g ; one group slab is contiguous
  const float4* p = (const float4*)(x + (size_t)bg * 8 * NN);
  float s = 0.f, q = 0.f;
  for (int i = threadIdx.x; i < 8192; i += 256) {
    float4 v = p[i];
    s += v.x + v.y + v.z + v.w;
    q += v.x * v.x + v.y * v.y + v.z * v.z + v.w * v.w;
  }
#pragma unroll
  for (int m = 1; m < 32; m <<= 1) {
    s += __shfl_xor(s, m, 32);
    q += __shfl_xor(q, m, 32);
  }
  __shared__ float ps[8], pq[8];
  const int wave = threadIdx.x >> 5, lane = threadIdx.x & 31;
  if (lane == 0) { ps[wave] = s; pq[wave] = q; }
  __syncthreads();
  if (threadIdx.x == 0) {
    float S = 0.f, Q = 0.f;
#pragma unroll
    for (int w = 0; w < 8; ++w) { S += ps[w]; Q += pq[w]; }
    const float mean = S * (1.f / 32768.f);
    const float var = Q * (1.f / 32768.f) - mean * mean;
    stats[bg * 2 + 0] = mean;
    stats[bg * 2 + 1] = rsqrtf(var + 1e-5f);
  }
}

// Normalize + transpose x[b][c][n] -> hn[b][n][c] (f16) via 64x64 LDS tile.
__global__ void __launch_bounds__(256) gn_apply_kernel(
    const float* __restrict__ x, const float* __restrict__ stats,
    const float* __restrict__ gamma, const float* __restrict__ beta,
    _Float16* __restrict__ hn) {
  __shared__ _Float16 tile[64 * 72];
  const int tid = threadIdx.x;
  const int b = blockIdx.z;
  const int c0 = blockIdx.y * 64;
  const int n0 = blockIdx.x * 64;
#pragma unroll
  for (int i = 0; i < 16; ++i) {
    const int cl = (tid >> 6) + i * 4;
    const int nl = tid & 63;
    const int c = c0 + cl;
    const float mean = stats[(b * 32 + (c >> 3)) * 2 + 0];
    const float rstd = stats[(b * 32 + (c >> 3)) * 2 + 1];
    const float v = x[((size_t)b * CC + c) * NN + n0 + nl];
    tile[nl * 72 + cl] = (_Float16)((v - mean) * rstd * gamma[c] + beta[c]);
  }
  __syncthreads();
#pragma unroll
  for (int i = 0; i < 16; ++i) {
    const int nl = (tid >> 6) + i * 4;
    const int cl = tid & 63;
    hn[((size_t)b * NN + n0 + nl) * CC + c0 + cl] = tile[nl * 72 + cl];
  }
}

// ------------------------------------------- weights f32 -> f16, done once
__global__ void __launch_bounds__(256) wcvt_kernel(
    const float* __restrict__ w0, const float* __restrict__ w1,
    const float* __restrict__ w2, const float* __restrict__ w3,
    _Float16* __restrict__ dst) {
  const int gid = blockIdx.x * 1024 + threadIdx.x * 4;
  const int m = gid >> 16;
  const int off = gid & 0xffff;
  const float* src = (m == 0) ? w0 : (m == 1) ? w1 : (m == 2) ? w2 : w3;
  float4 v = *(const float4*)(src + off);
  _Float16* d = dst + gid;
  d[0] = (_Float16)v.x; d[1] = (_Float16)v.y;
  d[2] = (_Float16)v.z; d[3] = (_Float16)v.w;
}

// ----------------------------------------------- Q / K^T GEMM (D[n][c] f16)
__global__ void __launch_bounds__(256) gemm_qk_kernel(
    const _Float16* __restrict__ hn, const _Float16* __restrict__ wq16,
    const float* __restrict__ bq, const _Float16* __restrict__ wk16,
    const float* __restrict__ bk, _Float16* __restrict__ Qm,
    _Float16* __restrict__ Km) {
  __shared__ _Float16 As[128 * 40];  // activations [row n][k]
  __shared__ _Float16 Bs[256 * 40];  // weights     [col o][k]
  const int tid = threadIdx.x, lane = tid & 31, wave = tid >> 5;
  const int b = blockIdx.y;
  const int n0 = blockIdx.x * 128;
  const size_t hbase = ((size_t)b * NN + n0) * CC;

  for (int wsel = 0; wsel < 2; ++wsel) {
    const _Float16* W = wsel ? wk16 : wq16;
    const float* bias = wsel ? bk : bq;
    _Float16* out = wsel ? Km : Qm;
    v8f acc[16];
#pragma unroll
    for (int t = 0; t < 16; ++t)
#pragma unroll
      for (int i = 0; i < 8; ++i) acc[t][i] = 0.f;

    for (int kc = 0; kc < 8; ++kc) {
      __syncthreads();
#pragma unroll
      for (int u = 0; u < 2; ++u) {  // 128x32 halves of hn
        const int idx = tid + 256 * u;
        const int r = idx >> 2, j = (idx & 3) * 8;
        *(v8h*)(As + r * 40 + j) =
            *(const v8h*)(hn + hbase + (size_t)r * CC + kc * 32 + j);
      }
#pragma unroll
      for (int u = 0; u < 4; ++u) {  // 256x32 halves of weights
        const int idx = tid + 256 * u;
        const int r = idx >> 2, j = (idx & 3) * 8;
        if (kc < 7)
          __builtin_prefetch(W + (size_t)r * CC + (kc + 1) * 32 + j, 0, 1);
        *(v8h*)(Bs + r * 40 + j) =
            *(const v8h*)(W + (size_t)r * CC + kc * 32 + j);
      }
      __syncthreads();
      const v16h a = frag_a(As, 40, wave * 16, 0, lane);
#pragma unroll
      for (int ct = 0; ct < 16; ++ct) {
        const v16h bf = frag_b(Bs, 40, ct * 16, 0, lane);
        acc[ct] = wmma16(a, bf, acc[ct]);
      }
    }
    const float qscale = (wsel == 0) ? 0.0625f : 1.0f;  // C^-0.5 folded in
    const int h = lane >> 4, l15 = lane & 15;
#pragma unroll
    for (int ct = 0; ct < 16; ++ct) {
      const int col = ct * 16 + l15;
      const float bcol = bias[col];
#pragma unroll
      for (int r = 0; r < 8; ++r) {
        const int rrow = wave * 16 + r + 8 * h;
        out[((size_t)b * NN + n0 + rrow) * CC + col] =
            (_Float16)((acc[ct][r] + bcol) * qscale);
      }
    }
  }
}

// ----------------------------------------- V GEMM (D[c][n] f16, A = weights)
__global__ void __launch_bounds__(256) gemm_v_kernel(
    const _Float16* __restrict__ hn, const _Float16* __restrict__ wv16,
    const float* __restrict__ bv, _Float16* __restrict__ Vm) {
  __shared__ _Float16 As[128 * 40];  // weights [row o][k]
  __shared__ _Float16 Bs[128 * 40];  // activations [col n][k]
  const int tid = threadIdx.x, lane = tid & 31, wave = tid >> 5;
  const int b = blockIdx.z;
  const int o0 = blockIdx.y * 128;
  const int n0 = blockIdx.x * 128;
  v8f acc[8];
#pragma unroll
  for (int t = 0; t < 8; ++t)
#pragma unroll
    for (int i = 0; i < 8; ++i) acc[t][i] = 0.f;

  for (int kc = 0; kc < 8; ++kc) {
    __syncthreads();
#pragma unroll
    for (int u = 0; u < 2; ++u) {
      const int idx = tid + 256 * u;
      const int r = idx >> 2, j = (idx & 3) * 8;
      *(v8h*)(As + r * 40 + j) =
          *(const v8h*)(wv16 + (size_t)(o0 + r) * CC + kc * 32 + j);
    }
#pragma unroll
    for (int u = 0; u < 2; ++u) {
      const int idx = tid + 256 * u;
      const int r = idx >> 2, j = (idx & 3) * 8;
      *(v8h*)(Bs + r * 40 + j) =
          *(const v8h*)(hn + ((size_t)b * NN + n0 + r) * CC + kc * 32 + j);
    }
    __syncthreads();
    const v16h a = frag_a(As, 40, wave * 16, 0, lane);
#pragma unroll
    for (int ct = 0; ct < 8; ++ct) {
      const v16h bf = frag_b(Bs, 40, ct * 16, 0, lane);
      acc[ct] = wmma16(a, bf, acc[ct]);
    }
  }
  const int h = lane >> 4, l15 = lane & 15;
#pragma unroll
  for (int r = 0; r < 8; ++r) {
    const int rrow = o0 + wave * 16 + r + 8 * h;
    const float brow = bv[rrow];
#pragma unroll
    for (int ct = 0; ct < 8; ++ct) {
      Vm[((size_t)b * CC + rrow) * NN + n0 + ct * 16 + l15] =
          (_Float16)(acc[ct][r] + brow);
    }
  }
}

// ------------------------------------------------------- Flash attention
// block: 128 queries, 8 waves (16-query strip each), keys streamed in
// 64-chunks; K/V chunks double-buffered via the Tensor Data Mover.
// LDS layout (bytes):
//   Qs  @ 0       : 128 x 264 f16
//   Ks  @ 67584   : 2 x (64 x 264) f16   (stride 33792)
//   Vs  @ 135168  : 2 x (256 x 72) f16   (stride 36864)
//   Ps  @ 208896  : 8 waves x 16 x 72 f16
#define QS_OFF   0u
#define KS_OFF   67584u
#define KS_STR   33792u
#define VS_OFF   135168u
#define VS_STR   36864u
#define PS_OFF   208896u
#define FLASH_LDS_BYTES 227328

__global__ void __launch_bounds__(256) flash_attn_kernel(
    const _Float16* __restrict__ Q, const _Float16* __restrict__ Kt,
    const _Float16* __restrict__ V, _Float16* __restrict__ AO) {
  extern __shared__ char smem[];
  _Float16* Qs = (_Float16*)(smem + QS_OFF);
  _Float16* Ps = (_Float16*)(smem + PS_OFF);

  const int tid = threadIdx.x, lane = tid & 31, wave = tid >> 5;
  const int h = lane >> 4, l15 = lane & 15;
  const int b = blockIdx.y;
  const int n0 = blockIdx.x * 128;
  const int row0 = wave * 16;
  _Float16* ps = Ps + wave * 16 * 72;
  const _Float16* Qg = Q + ((size_t)b * NN + n0) * CC;
  const _Float16* Kg = Kt + (size_t)b * NN * CC;
  const _Float16* Vg = V + (size_t)b * CC * NN;

#if HAVE_TDM
  if (wave == 0) {  // TDM ignores EXEC: must be wave-granular branch
    tdm_load_2d(QS_OFF, Qg, 256, 128, 256, 6, 3);        // pitch 264
    tdm_load_2d(KS_OFF, Kg, 256, 64, 256, 6, 3);         // pitch 264
    tdm_load_2d(VS_OFF, Vg, 64, 256, NN, 4, 3);          // pitch 72
  }
#else
#pragma unroll
  for (int u = 0; u < 16; ++u) {
    const int idx = tid + 256 * u;
    const int r = idx >> 5, j = (idx & 31) * 8;
    *(v8h*)(Qs + r * 264 + j) = *(const v8h*)(Qg + (size_t)r * CC + j);
  }
#endif

  v8f o[16];
#pragma unroll
  for (int t = 0; t < 16; ++t)
#pragma unroll
    for (int i = 0; i < 8; ++i) o[t][i] = 0.f;
  float m8[8], l8[8];
#pragma unroll
  for (int r = 0; r < 8; ++r) { m8[r] = -1e30f; l8[r] = 0.f; }

  for (int it = 0; it < NN / 64; ++it) {
    const unsigned cur = (unsigned)(it & 1);
    const int key0 = it * 64;
    // double-buffer pointers computed from byte offsets (no pointer arrays:
    // LDS addrspacecast in static initializers is not linkable)
    _Float16* Ks = (_Float16*)(smem + KS_OFF + cur * KS_STR);
    _Float16* Vs = (_Float16*)(smem + VS_OFF + cur * VS_STR);
#if HAVE_TDM
    if (wave == 0) {
      if (it + 1 < NN / 64) {  // prefetch next chunk into the other buffer
        const unsigned nxt = cur ^ 1u;
        tdm_load_2d(KS_OFF + nxt * KS_STR, Kg + (size_t)(key0 + 64) * CC, 256,
                    64, 256, 6, 3);
        tdm_load_2d(VS_OFF + nxt * VS_STR, Vg + key0 + 64, 64, 256, NN, 4, 3);
        __builtin_amdgcn_s_wait_tensorcnt(2);  // current chunk landed
      } else {
        __builtin_amdgcn_s_wait_tensorcnt(0);
      }
    }
    __syncthreads();
#else
    __syncthreads();
#pragma unroll
    for (int u = 0; u < 8; ++u) {
      const int idx = tid + 256 * u;
      const int r = idx >> 5, j = (idx & 31) * 8;
      *(v8h*)(Ks + r * 264 + j) =
          *(const v8h*)(Kg + (size_t)(key0 + r) * CC + j);
    }
#pragma unroll
    for (int u = 0; u < 8; ++u) {
      const int idx = tid + 256 * u;
      const int r = idx >> 3, j = (idx & 7) * 8;
      *(v8h*)(Vs + r * 72 + j) = *(const v8h*)(Vg + (size_t)r * NN + key0 + j);
    }
    __syncthreads();
#endif

    // S strip = Q_strip x K_chunk^T  (16 x 64)
    v8f s[4];
#pragma unroll
    for (int t = 0; t < 4; ++t)
#pragma unroll
      for (int i = 0; i < 8; ++i) s[t][i] = 0.f;
    for (int kc = 0; kc < 8; ++kc) {
      const v16h a = frag_a(Qs, 264, row0, kc * 32, lane);
#pragma unroll
      for (int ct = 0; ct < 4; ++ct) {
        const v16h bf = frag_b(Ks, 264, ct * 16, kc * 32, lane);
        s[ct] = wmma16(a, bf, s[ct]);
      }
    }

    // online softmax per row (row = r + 8*h)
#pragma unroll
    for (int r = 0; r < 8; ++r) {
      float lm = -1e30f;
#pragma unroll
      for (int ct = 0; ct < 4; ++ct) lm = fmaxf(lm, s[ct][r]);
      lm = rmax16(lm);
      const float mnew = fmaxf(m8[r], lm);
      const float corr = __expf(m8[r] - mnew);
      float rs = 0.f;
#pragma unroll
      for (int ct = 0; ct < 4; ++ct) {
        const float p = __expf(s[ct][r] - mnew);
        ps[(r + 8 * h) * 72 + ct * 16 + l15] = (_Float16)p;
        rs += p;
      }
      rs = rsum16(rs);
      l8[r] = l8[r] * corr + rs;
      m8[r] = mnew;
#pragma unroll
      for (int co = 0; co < 16; ++co) o[co][r] *= corr;
    }

    // O += P x V_chunk^T  (K-dim = 64 keys)
#pragma unroll
    for (int kc2 = 0; kc2 < 2; ++kc2) {
      const v16h a = frag_a(ps, 72, 0, kc2 * 32, lane);
#pragma unroll
      for (int co = 0; co < 16; ++co) {
        const v16h bf = frag_b(Vs, 72, co * 16, kc2 * 32, lane);
        o[co] = wmma16(a, bf, o[co]);
      }
    }
    __syncthreads();  // consumers done before DMA overwrites this buffer
  }

  float inv[8];
#pragma unroll
  for (int r = 0; r < 8; ++r) inv[r] = 1.f / l8[r];
#pragma unroll
  for (int co = 0; co < 16; ++co)
#pragma unroll
    for (int r = 0; r < 8; ++r) {
      const int rrow = row0 + r + 8 * h;
      AO[((size_t)b * NN + n0 + rrow) * CC + co * 16 + l15] =
          (_Float16)(o[co][r] * inv[r]);
    }
}

// ---------------------------- Output projection + residual (f32, coalesced)
__global__ void __launch_bounds__(256) gemm_proj_kernel(
    const _Float16* __restrict__ ao, const _Float16* __restrict__ wp16,
    const float* __restrict__ bp, const float* __restrict__ xres,
    float* __restrict__ out) {
  __shared__ _Float16 As[128 * 40];  // wp rows o
  __shared__ _Float16 Bs[128 * 40];  // ao rows n
  const int tid = threadIdx.x, lane = tid & 31, wave = tid >> 5;
  const int b = blockIdx.z;
  const int o0 = blockIdx.y * 128;
  const int n0 = blockIdx.x * 128;
  v8f acc[8];
#pragma unroll
  for (int t = 0; t < 8; ++t)
#pragma unroll
    for (int i = 0; i < 8; ++i) acc[t][i] = 0.f;

  for (int kc = 0; kc < 8; ++kc) {
    __syncthreads();
#pragma unroll
    for (int u = 0; u < 2; ++u) {
      const int idx = tid + 256 * u;
      const int r = idx >> 2, j = (idx & 3) * 8;
      *(v8h*)(As + r * 40 + j) =
          *(const v8h*)(wp16 + (size_t)(o0 + r) * CC + kc * 32 + j);
    }
#pragma unroll
    for (int u = 0; u < 2; ++u) {
      const int idx = tid + 256 * u;
      const int r = idx >> 2, j = (idx & 3) * 8;
      *(v8h*)(Bs + r * 40 + j) =
          *(const v8h*)(ao + ((size_t)b * NN + n0 + r) * CC + kc * 32 + j);
    }
    __syncthreads();
    const v16h a = frag_a(As, 40, wave * 16, 0, lane);
#pragma unroll
    for (int ct = 0; ct < 8; ++ct) {
      const v16h bf = frag_b(Bs, 40, ct * 16, 0, lane);
      acc[ct] = wmma16(a, bf, acc[ct]);
    }
  }
  const int h = lane >> 4, l15 = lane & 15;
#pragma unroll
  for (int r = 0; r < 8; ++r) {
    const int rrow = o0 + wave * 16 + r + 8 * h;
    const float brow = bp[rrow];
#pragma unroll
    for (int ct = 0; ct < 8; ++ct) {
      const size_t idx = ((size_t)b * CC + rrow) * NN + n0 + ct * 16 + l15;
      out[idx] = xres[idx] + acc[ct][r] + brow;
    }
  }
}

// ---------------------------------------------------------------- launcher
extern "C" void kernel_launch(void* const* d_in, const int* in_sizes, int n_in,
                              void* d_out, int out_size, void* d_ws,
                              size_t ws_size, hipStream_t stream) {
  (void)n_in; (void)out_size; (void)ws_size;
  const float* x    = (const float*)d_in[0];
  const float* temb = (const float*)d_in[1];
  const float* gn_w = (const float*)d_in[2];
  const float* gn_b = (const float*)d_in[3];
  const float* wq   = (const float*)d_in[4];
  const float* bq   = (const float*)d_in[5];
  const float* wk   = (const float*)d_in[6];
  const float* bk   = (const float*)d_in[7];
  const float* wv   = (const float*)d_in[8];
  const float* bv   = (const float*)d_in[9];
  const float* wp   = (const float*)d_in[10];
  const float* bp   = (const float*)d_in[11];
  float* out = (float*)d_out;

  const size_t MAT = (size_t)BB * NN * CC;  // 8.39M elements
  float* stats = (float*)d_ws;
  _Float16* hn = (_Float16*)((char*)d_ws + 4096);
  _Float16* Qm = hn + MAT;
  _Float16* Km = Qm + MAT;
  _Float16* Vm = Km + MAT;
  _Float16* AO = Vm + MAT;
  _Float16* Wh = AO + MAT;  // 4 x 256 x 256 f16
  _Float16* wq16 = Wh + 0 * 65536;
  _Float16* wk16 = Wh + 1 * 65536;
  _Float16* wv16 = Wh + 2 * 65536;
  _Float16* wp16 = Wh + 3 * 65536;

  wcvt_kernel<<<dim3(256), 256, 0, stream>>>(wq, wk, wv, wp, Wh);
  gn_stats_kernel<<<dim3(BB * 32), 256, 0, stream>>>(x, stats);
  gn_apply_kernel<<<dim3(NN / 64, CC / 64, BB), 256, 0, stream>>>(
      x, stats, gn_w, gn_b, hn);
  gemm_qk_kernel<<<dim3(NN / 128, BB), 256, 0, stream>>>(hn, wq16, bq, wk16,
                                                         bk, Qm, Km);
  gemm_v_kernel<<<dim3(NN / 128, CC / 128, BB), 256, 0, stream>>>(hn, wv16, bv,
                                                                  Vm);
  (void)hipFuncSetAttribute((const void*)flash_attn_kernel,
                            hipFuncAttributeMaxDynamicSharedMemorySize,
                            FLASH_LDS_BYTES);
  flash_attn_kernel<<<dim3(NN / 128, BB), 256, FLASH_LDS_BYTES, stream>>>(
      Qm, Km, Vm, AO);
  gemm_proj_kernel<<<dim3(NN / 128, CC / 128, BB), 256, 0, stream>>>(AO, wp16,
                                                                     bp, x, out);
  (void)hipMemcpyAsync(out + MAT, temb, (size_t)in_sizes[1] * sizeof(float),
                       hipMemcpyDeviceToDevice, stream);
}
